// GRUHANClassifier_15410342658116
// MI455X (gfx1250) — matrix operand order repaired and checked by reading, not compiled, not run on hardware
//
#include <hip/hip_runtime.h>
#include <hip/hip_bf16.h>

// ---------------------------------------------------------------------------
// GRU-HAN classifier for MI455X (gfx1250), wave32 + v_wmma_f32_16x16x32_bf16.
// ---------------------------------------------------------------------------

typedef __attribute__((ext_vector_type(16))) __bf16 v16bf;
typedef __attribute__((ext_vector_type(8)))  float  v8f;

// 16x32 bf16 A-fragment: per lane, elements 0..7  = K hi*8 + 0..7 (contiguous)
//                                  elements 8..15 = K 16 + hi*8 + 0..7 (contiguous)
// -> two 16-byte chunks per lane.
union FragA { v16bf v; uint4 q[2]; };

#define NB  32      // B
#define NP  64      // P
#define NT_ 128     // T
#define NE  100     // E
#define NH  100     // H
#define NPOSTS 2048 // B*P
#define EPAD_POST 128
#define EPAD_USER 224
#define HSTRIDE   224   // 2H=200 padded to 7*32

// ---------------------------------------------------------------------------
// 1) Pre-swizzle a weight matrix W[Nrows][Kdim] (f32, row-major; used as
//    B = W^T, K x N) into WMMA B-fragment chunks:
//    chunk(nt,ks) = 32 lanes x 16 bf16; element i of lane L is
//    (K = ks*32 + 8*(L/16) + i + (i>=8 ? 8 : 0), N = nt*16 + L%16), zero outside bounds.
// ---------------------------------------------------------------------------
__global__ void swizzle_b_kernel(const float* __restrict__ W,
                                 __hip_bfloat16* __restrict__ out,
                                 int Nrows, int Kdim, int NT, int KS) {
  int idx = blockIdx.x * blockDim.x + threadIdx.x;  // one thread per (chunk,lane)
  int total = NT * KS * 32;
  if (idx >= total) return;
  int lane  = idx & 31;
  int chunk = idx >> 5;          // nt*KS + ks
  int ks = chunk % KS;
  int nt = chunk / KS;
  int n = nt * 16 + (lane & 15);
  int kbase = ks * 32 + ((lane >> 4) << 3);
  __hip_bfloat16* o = out + (size_t)chunk * 32 * 16 + (size_t)lane * 16;
  for (int i = 0; i < 16; ++i) {
    int k = kbase + i + (i >= 8 ? 8 : 0);
    float v = (n < Nrows && k < Kdim) ? W[(size_t)n * Kdim + k] : 0.f;
    o[i] = __float2bfloat16(v);
  }
}

// ---------------------------------------------------------------------------
// 2) Embedding gather: X[n][t][e] bf16, e in [0,128), zero-padded.
// ---------------------------------------------------------------------------
__global__ void gather_embed_kernel(const int* __restrict__ ids,
                                    const float* __restrict__ emb,
                                    __hip_bfloat16* __restrict__ X) {
  size_t idx = (size_t)blockIdx.x * blockDim.x + threadIdx.x;
  if (idx >= (size_t)NPOSTS * NT_ * EPAD_POST) return;
  int e = (int)(idx & (EPAD_POST - 1));
  size_t nt = idx >> 7;  // n*T + t
  int id = ids[nt];
  float v = (e < NE && id != 0) ? emb[(size_t)id * NE + e] : 0.f;
  X[idx] = __float2bfloat16(v);
}

// ---------------------------------------------------------------------------
// 3) Per-post validity mask for the user level (token count > 2).
// ---------------------------------------------------------------------------
__global__ void postmask_kernel(const int* __restrict__ ids,
                                int* __restrict__ pmask) {
  int p = blockIdx.x * blockDim.x + threadIdx.x;
  if (p >= NPOSTS) return;
  int c = 0;
  for (int t = 0; t < NT_; ++t) c += (ids[(size_t)p * NT_ + t] != 0);
  pmask[p] = (c > 2) ? 1 : 0;
}

// ---------------------------------------------------------------------------
// 4) One direction of a GRU layer. Block = 16 sequences, 8 waves.
//    gates[16,300(+pad)] = x_t @ Wih^T  and  h @ Whh^T  via WMMA bf16,
//    then fused PyTorch-GRU-cell update with masked carry.
//    HS output: bf16, stride HSTRIDE, column offset dir*NH.
// ---------------------------------------------------------------------------
__global__ __launch_bounds__(256) void gru_dir_kernel(
    const __hip_bfloat16* __restrict__ X, int EPAD, int KSX,
    const __hip_bfloat16* __restrict__ Wih,   // swizzled, 19*KSX chunks
    const __hip_bfloat16* __restrict__ Whh,   // swizzled, 19*4   chunks
    const float* __restrict__ bih, const float* __restrict__ bhh,
    const int* __restrict__ step_mask, int T, int dir,
    __hip_bfloat16* __restrict__ HS) {
  __shared__ float gx_s[16][304];
  __shared__ float gh_s[16][304];
  __shared__ float hf_s[16][104];
  __shared__ __align__(16) __hip_bfloat16 hb_s[16][128];
  __shared__ int mv_s[16];

  const int tid  = threadIdx.x;
  const int lane = tid & 31;
  const int wave = tid >> 5;
  const int row  = lane & 15;
  const int hi   = lane >> 4;
  const int mbase = blockIdx.x * 16;

  for (int e = tid; e < 16 * 104; e += 256) (&hf_s[0][0])[e] = 0.f;
  for (int e = tid; e < 16 * 128; e += 256) (&hb_s[0][0])[e] = __float2bfloat16(0.f);
  __syncthreads();

  for (int t = 0; t < T; ++t) {
    const int tt = dir ? (T - 1 - t) : t;

    // Stage this step's carry mask (visible after the mid-step barrier).
    if (tid < 16) mv_s[tid] = step_mask[(size_t)(mbase + tid) * T + tt];

    // Build A fragments: two b128 chunks per K-slice per lane.
    FragA ax[7];
    FragA ah[4];
    const uint4* xq =
        (const uint4*)(const void*)(X + ((size_t)(mbase + row) * T + tt) * EPAD);
    for (int ks = 0; ks < KSX; ++ks) {
      ax[ks].q[0] = xq[ks * 4 + hi];
      ax[ks].q[1] = xq[ks * 4 + hi + 2];
    }
    const uint4* hq = (const uint4*)(const void*)(&hb_s[row][0]);
    for (int ks = 0; ks < 4; ++ks) {
      ah[ks].q[0] = hq[ks * 4 + hi];
      ah[ks].q[1] = hq[ks * 4 + hi + 2];
    }

    // Each wave owns N-tiles {wave, wave+8, wave+16} of 19.
    for (int nt = wave; nt < 19; nt += 8) {
      v8f accx = {0.f, 0.f, 0.f, 0.f, 0.f, 0.f, 0.f, 0.f};
      v8f acch = {0.f, 0.f, 0.f, 0.f, 0.f, 0.f, 0.f, 0.f};
      const v16bf* bih_p = (const v16bf*)(const void*)Wih + (size_t)nt * KSX * 32 + lane;
      const v16bf* bhh_p = (const v16bf*)(const void*)Whh + (size_t)nt * 4 * 32 + lane;
      for (int ks = 0; ks < KSX; ++ks)
        accx = __builtin_amdgcn_wmma_f32_16x16x32_bf16(
            false, ax[ks].v, false, bih_p[ks * 32], (short)0, accx, false, false);
      for (int ks = 0; ks < 4; ++ks)
        acch = __builtin_amdgcn_wmma_f32_16x16x32_bf16(
            false, ah[ks].v, false, bhh_p[ks * 32], (short)0, acch, false, false);
      const int col = nt * 16 + row;
      for (int r = 0; r < 8; ++r) {
        int m = r + (hi << 3);
        gx_s[m][col] = accx[r];
        gh_s[m][col] = acch[r];
      }
    }
    __syncthreads();

    // Fused GRU cell update over 16 x 100 hidden elements.
    for (int e = tid; e < 16 * NH; e += 256) {
      int m = e / NH, j = e - m * NH;
      float rg = gx_s[m][j] + bih[j] + gh_s[m][j] + bhh[j];
      float zg = gx_s[m][NH + j] + bih[NH + j] + gh_s[m][NH + j] + bhh[NH + j];
      float ngx = gx_s[m][2 * NH + j] + bih[2 * NH + j];
      float ngh = gh_s[m][2 * NH + j] + bhh[2 * NH + j];
      float rv = 1.f / (1.f + __expf(-rg));
      float zv = 1.f / (1.f + __expf(-zg));
      float nv = tanhf(ngx + rv * ngh);
      float hold = hf_s[m][j];
      float hnew = (1.f - zv) * nv + zv * hold;
      float h = (mv_s[m] != 0) ? hnew : hold;
      hf_s[m][j] = h;
      hb_s[m][j] = __float2bfloat16(h);
      HS[((size_t)(mbase + m) * T + tt) * HSTRIDE + dir * NH + j] = __float2bfloat16(h);
    }
    __syncthreads();
  }
}

// ---------------------------------------------------------------------------
// 5) Attention pooling for one sequence per block:
//    att[t] = sum_a tanh(OUT[t]·w_att[a] + b_att[a]) * w_ctx[a]  (WMMA GEMM),
//    masked softmax, pooled[d] = sum_t alpha[t]*OUT[t][d].
// ---------------------------------------------------------------------------
__global__ __launch_bounds__(256) void attn_kernel(
    const __hip_bfloat16* __restrict__ HS, int Tn,
    const __hip_bfloat16* __restrict__ Watt,  // swizzled 7x7 chunks
    const float* __restrict__ batt, const float* __restrict__ wctx,
    const int* __restrict__ step_mask,
    __hip_bfloat16* __restrict__ pooled_bf,   // stride HSTRIDE (or null)
    float* __restrict__ pooled_f,             // stride 200 (or null)
    float* __restrict__ alphas_out) {         // stride Tn (or null)
  __shared__ float att_s[128];
  __shared__ float red_s[2];
  const int tid  = threadIdx.x;
  const int lane = tid & 31;
  const int wave = tid >> 5;
  const int row  = lane & 15;
  const int hi   = lane >> 4;
  const int n = blockIdx.x;
  const int mtiles = Tn >> 4;

  if (wave < mtiles) {
    const int trow = wave * 16 + row;  // timestep row of A
    const uint4* aq =
        (const uint4*)(const void*)(HS + ((size_t)n * Tn + trow) * HSTRIDE);
    FragA af[7];
    for (int ks = 0; ks < 7; ++ks) {
      af[ks].q[0] = aq[ks * 4 + hi];
      af[ks].q[1] = aq[ks * 4 + hi + 2];
    }
    float ar[8] = {0.f, 0.f, 0.f, 0.f, 0.f, 0.f, 0.f, 0.f};
    for (int nt = 0; nt < 7; ++nt) {
      v8f c = {0.f, 0.f, 0.f, 0.f, 0.f, 0.f, 0.f, 0.f};
      const v16bf* bp = (const v16bf*)(const void*)Watt + (size_t)nt * 7 * 32 + lane;
      for (int ks = 0; ks < 7; ++ks)
        c = __builtin_amdgcn_wmma_f32_16x16x32_bf16(
            false, af[ks].v, false, bp[ks * 32], (short)0, c, false, false);
      int col = nt * 16 + row;
      if (col < 100) {
        float ba = batt[col], wc = wctx[col];
        for (int r = 0; r < 8; ++r) ar[r] += tanhf(c[r] + ba) * wc;
      }
    }
    for (int r = 0; r < 8; ++r) {
      float v = ar[r];
      v += __shfl_xor(v, 1, 16);
      v += __shfl_xor(v, 2, 16);
      v += __shfl_xor(v, 4, 16);
      v += __shfl_xor(v, 8, 16);
      if (row == 0) att_s[wave * 16 + r + (hi << 3)] = v;
    }
  }
  __syncthreads();

  if (tid == 0) {
    float mx = -1e30f;
    for (int t = 0; t < Tn; ++t)
      if (step_mask[(size_t)n * Tn + t] != 0 && att_s[t] > mx) mx = att_s[t];
    red_s[0] = mx;
  }
  __syncthreads();
  if (tid < Tn) {
    bool valid = step_mask[(size_t)n * Tn + tid] != 0;
    att_s[tid] = valid ? __expf(att_s[tid] - red_s[0]) : 0.f;
  }
  __syncthreads();
  if (tid == 0) {
    float s = 0.f;
    for (int t = 0; t < Tn; ++t) s += att_s[t];
    red_s[1] = (s > 0.f) ? 1.f / s : 0.f;
  }
  __syncthreads();
  if (tid < Tn) {
    att_s[tid] *= red_s[1];
    if (alphas_out) alphas_out[(size_t)n * Tn + tid] = att_s[tid];
  }
  __syncthreads();

  // Pooling: read bf16 pairs as dwords (d even, 4B aligned).
  for (int d2 = tid; d2 < NH; d2 += 256) {  // 100 pairs cover 200 dims
    int d = 2 * d2;
    float s0 = 0.f, s1 = 0.f;
    const unsigned* base =
        (const unsigned*)(const void*)(HS + (size_t)n * Tn * HSTRIDE + d);
    for (int t = 0; t < Tn; ++t) {
      unsigned pk = base[(size_t)t * (HSTRIDE / 2)];
      __hip_bfloat16 lo = __ushort_as_bfloat16((unsigned short)(pk & 0xFFFF));
      __hip_bfloat16 hi2 = __ushort_as_bfloat16((unsigned short)(pk >> 16));
      float a = att_s[t];
      s0 += a * __bfloat162float(lo);
      s1 += a * __bfloat162float(hi2);
    }
    if (pooled_bf) {
      pooled_bf[(size_t)n * HSTRIDE + d]     = __float2bfloat16(s0);
      pooled_bf[(size_t)n * HSTRIDE + d + 1] = __float2bfloat16(s1);
    }
    if (pooled_f) {
      pooled_f[(size_t)n * (2 * NH) + d]     = s0;
      pooled_f[(size_t)n * (2 * NH) + d + 1] = s1;
    }
  }
}

// ---------------------------------------------------------------------------
// 6) Classifier: logits[32,2] = feat @ w_clf^T + b_clf.
// ---------------------------------------------------------------------------
__global__ void clf_kernel(const float* __restrict__ feat,
                           const float* __restrict__ wclf,
                           const float* __restrict__ bclf,
                           float* __restrict__ out) {
  int tid = threadIdx.x;
  if (tid >= NB * 2) return;
  int b = tid >> 1, c = tid & 1;
  float s = bclf[c];
  for (int d = 0; d < 2 * NH; ++d) s += feat[b * 2 * NH + d] * wclf[c * 2 * NH + d];
  out[b * 2 + c] = s;
}

// ---------------------------------------------------------------------------
// Host launcher.
// ---------------------------------------------------------------------------
extern "C" void kernel_launch(void* const* d_in, const int* in_sizes, int n_in,
                              void* d_out, int out_size, void* d_ws, size_t ws_size,
                              hipStream_t stream) {
  (void)n_in; (void)out_size; (void)ws_size;
  const int*   ids = (const int*)d_in[0];
  const float* emb = (const float*)d_in[3];
  const int PBASE = 4, UBASE = 15;

  // Nested-dict flatten order: insertion vs alphabetical (detected via sizes).
  // insertion: wihf whhf bihf bhhf wihb whhb bihb bhhb watt batt wctx
  // sorted   : batt bhhb bhhf bihb bihf watt wctx whhb whhf wihb wihf
  int rel[11];
  if (in_sizes[PBASE] == 100) {  // sorted: first entry is b_att (100)
    rel[0] = 10; rel[1] = 8; rel[2] = 4; rel[3] = 2;   // wihf whhf bihf bhhf
    rel[4] = 9;  rel[5] = 7; rel[6] = 3; rel[7] = 1;   // wihb whhb bihb bhhb
    rel[8] = 5;  rel[9] = 0; rel[10] = 6;              // watt batt wctx
  } else {
    for (int i = 0; i < 11; ++i) rel[i] = i;
  }
  const float* p_wihf = (const float*)d_in[PBASE + rel[0]];
  const float* p_whhf = (const float*)d_in[PBASE + rel[1]];
  const float* p_bihf = (const float*)d_in[PBASE + rel[2]];
  const float* p_bhhf = (const float*)d_in[PBASE + rel[3]];
  const float* p_wihb = (const float*)d_in[PBASE + rel[4]];
  const float* p_whhb = (const float*)d_in[PBASE + rel[5]];
  const float* p_bihb = (const float*)d_in[PBASE + rel[6]];
  const float* p_bhhb = (const float*)d_in[PBASE + rel[7]];
  const float* p_watt = (const float*)d_in[PBASE + rel[8]];
  const float* p_batt = (const float*)d_in[PBASE + rel[9]];
  const float* p_wctx = (const float*)d_in[PBASE + rel[10]];
  const float* u_wihf = (const float*)d_in[UBASE + rel[0]];
  const float* u_whhf = (const float*)d_in[UBASE + rel[1]];
  const float* u_bihf = (const float*)d_in[UBASE + rel[2]];
  const float* u_bhhf = (const float*)d_in[UBASE + rel[3]];
  const float* u_wihb = (const float*)d_in[UBASE + rel[4]];
  const float* u_whhb = (const float*)d_in[UBASE + rel[5]];
  const float* u_bihb = (const float*)d_in[UBASE + rel[6]];
  const float* u_bhhb = (const float*)d_in[UBASE + rel[7]];
  const float* u_watt = (const float*)d_in[UBASE + rel[8]];
  const float* u_batt = (const float*)d_in[UBASE + rel[9]];
  const float* u_wctx = (const float*)d_in[UBASE + rel[10]];
  const float* w_clf  = (const float*)d_in[26];
  const float* b_clf  = (const float*)d_in[27];
  float* dout = (float*)d_out;

  // ---- workspace layout (all sizes 256B multiples) ----
  char* ws = (char*)d_ws;
  size_t off = 0;
  const size_t WX_B   = (size_t)NPOSTS * NT_ * EPAD_POST * 2;  // 64 MiB
  const size_t HSP_B  = (size_t)NPOSTS * NT_ * HSTRIDE * 2;    // 112 MiB
  const size_t XU_B   = (size_t)NB * NP * HSTRIDE * 2;
  const size_t HSU_B  = XU_B;
  const size_t FEAT_B = (size_t)NB * 2 * NH * 4;
  const size_t PM_B   = (size_t)NPOSTS * 4;
  const size_t SW4_B  = (size_t)19 * 4 * 1024;
  const size_t SW7_B  = (size_t)19 * 7 * 1024;
  const size_t SWA_B  = (size_t)7 * 7 * 1024;
  __hip_bfloat16* WX  = (__hip_bfloat16*)(ws + off); off += WX_B;
  __hip_bfloat16* HSP = (__hip_bfloat16*)(ws + off); off += HSP_B;
  __hip_bfloat16* XU  = (__hip_bfloat16*)(ws + off); off += XU_B;
  __hip_bfloat16* HSU = (__hip_bfloat16*)(ws + off); off += HSU_B;
  float* FEAT = (float*)(ws + off); off += FEAT_B;
  int*   PM   = (int*)(ws + off);   off += PM_B;
  __hip_bfloat16* SWP_IHF = (__hip_bfloat16*)(ws + off); off += SW4_B;
  __hip_bfloat16* SWP_HHF = (__hip_bfloat16*)(ws + off); off += SW4_B;
  __hip_bfloat16* SWP_IHB = (__hip_bfloat16*)(ws + off); off += SW4_B;
  __hip_bfloat16* SWP_HHB = (__hip_bfloat16*)(ws + off); off += SW4_B;
  __hip_bfloat16* SWU_IHF = (__hip_bfloat16*)(ws + off); off += SW7_B;
  __hip_bfloat16* SWU_IHB = (__hip_bfloat16*)(ws + off); off += SW7_B;
  __hip_bfloat16* SWU_HHF = (__hip_bfloat16*)(ws + off); off += SW4_B;
  __hip_bfloat16* SWU_HHB = (__hip_bfloat16*)(ws + off); off += SW4_B;
  __hip_bfloat16* SWATT_P = (__hip_bfloat16*)(ws + off); off += SWA_B;
  __hip_bfloat16* SWATT_U = (__hip_bfloat16*)(ws + off); off += SWA_B;

  // Zero the padded hidden-state / pooled buffers (pad columns must be 0).
  hipMemsetAsync(HSP, 0, HSP_B, stream);
  hipMemsetAsync(XU, 0, XU_B, stream);
  hipMemsetAsync(HSU, 0, HSU_B, stream);

  // ---- weight swizzles ----
  auto sw = [&](const float* W, __hip_bfloat16* o, int Nr, int Kd, int NTt, int KSs) {
    int total = NTt * KSs * 32;
    swizzle_b_kernel<<<(total + 255) / 256, 256, 0, stream>>>(W, o, Nr, Kd, NTt, KSs);
  };
  sw(p_wihf, SWP_IHF, 300, 100, 19, 4);
  sw(p_whhf, SWP_HHF, 300, 100, 19, 4);
  sw(p_wihb, SWP_IHB, 300, 100, 19, 4);
  sw(p_whhb, SWP_HHB, 300, 100, 19, 4);
  sw(u_wihf, SWU_IHF, 300, 200, 19, 7);
  sw(u_wihb, SWU_IHB, 300, 200, 19, 7);
  sw(u_whhf, SWU_HHF, 300, 100, 19, 4);
  sw(u_whhb, SWU_HHB, 300, 100, 19, 4);
  sw(p_watt, SWATT_P, 100, 200, 7, 7);
  sw(u_watt, SWATT_U, 100, 200, 7, 7);

  // ---- embedding gather + post mask ----
  {
    size_t total = (size_t)NPOSTS * NT_ * EPAD_POST;
    gather_embed_kernel<<<(unsigned)((total + 255) / 256), 256, 0, stream>>>(ids, emb, WX);
    postmask_kernel<<<(NPOSTS + 255) / 256, 256, 0, stream>>>(ids, PM);
  }

  // ---- post-level bi-GRU (mask = ids != 0) ----
  gru_dir_kernel<<<NPOSTS / 16, 256, 0, stream>>>(
      WX, EPAD_POST, 4, SWP_IHF, SWP_HHF, p_bihf, p_bhhf, ids, NT_, 0, HSP);
  gru_dir_kernel<<<NPOSTS / 16, 256, 0, stream>>>(
      WX, EPAD_POST, 4, SWP_IHB, SWP_HHB, p_bihb, p_bhhb, ids, NT_, 1, HSP);

  // ---- post-level attention pooling -> user-level inputs (bf16, stride 224)
  attn_kernel<<<NPOSTS, 256, 0, stream>>>(
      HSP, NT_, SWATT_P, p_batt, p_wctx, ids, XU, nullptr, nullptr);

  // ---- user-level bi-GRU (N=32, T=64, input 200) ----
  gru_dir_kernel<<<NB / 16, 256, 0, stream>>>(
      XU, EPAD_USER, 7, SWU_IHF, SWU_HHF, u_bihf, u_bhhf, PM, NP, 0, HSU);
  gru_dir_kernel<<<NB / 16, 256, 0, stream>>>(
      XU, EPAD_USER, 7, SWU_IHB, SWU_HHB, u_bihb, u_bhhb, PM, NP, 1, HSU);

  // ---- user-level attention: feat + alphas (alphas -> d_out[64..2111]) ----
  attn_kernel<<<NB, 256, 0, stream>>>(
      HSU, NP, SWATT_U, u_batt, u_wctx, PM, nullptr, FEAT, dout + NB * 2);

  // ---- classifier: logits -> d_out[0..63] ----
  clf_kernel<<<1, 64, 0, stream>>>(FEAT, w_clf, b_clf, dout);
}
